// MoE_20839181320263
// MI455X (gfx1250) — compile-verified, matrix operand order checked by reading, not run on hardware
//
#include <hip/hip_runtime.h>
#include <hip/hip_bf16.h>

// ---------------------------------------------------------------------------
// MoE (top-2 of 8 experts, SwiGLU) for MI455X / gfx1250, wave32 + WMMA bf16.
//
// Roofline: 402MB fp32 weights -> one-time bf16 transpose-convert (201MB).
// Expert GEMMs are L2-bandwidth bound; M-tile=128 so every weight byte
// streamed from L2 feeds 128 output rows. v_wmma_f32_16x16x32_bf16 with f32
// accumulate. Deterministic: each token row's result is independent of its
// position in the gathered tile; final combine sums exactly its two slots.
// ---------------------------------------------------------------------------

#define TKN   8192     // B*S tokens
#define DM    1024     // model dim
#define HM    4096     // hidden dim
#define EM    8        // experts
#define CAP   8192     // worst-case tokens per expert
#define MT    128      // token tile (gathered rows) -> 8 WMMA M-subtiles
#define MSH   6        // log2(CAP/MT) tiles per expert
#define NT    128      // N tile per workgroup (8 waves x 16)
#define KT    32       // K step (WMMA bf16 K)
#define LDA   40       // LDS row pitch in bf16 (80B -> conflict-friendly)

#if __has_builtin(__builtin_amdgcn_load_to_lds) && __has_builtin(__builtin_amdgcn_s_wait_asynccnt)
#define USE_ASYNC_LDS 1
#endif

typedef __attribute__((ext_vector_type(16))) __bf16 v16bf;
typedef __attribute__((ext_vector_type(8)))  float  v8f;

union FragAB { v16bf v; uint4 q[2]; };

static __device__ inline __bf16 to_bf16(float f) {
    unsigned u = __builtin_bit_cast(unsigned, f);
    unsigned r = (u + 0x7FFFu + ((u >> 16) & 1u)) >> 16;   // RNE
    return __builtin_bit_cast(__bf16, (unsigned short)r);
}

static __device__ inline v8f wmma_bf16(v16bf a, v16bf b, v8f c) {
    // D = A(16x32) * B(32x16) + C, f32 accumulate
    return __builtin_amdgcn_wmma_f32_16x16x32_bf16(
        /*neg_a=*/false, a, /*neg_b=*/false, b,
        /*c_mod=*/(short)0, c, /*reuse_a=*/false, /*reuse_b=*/false);
}

// ---------------------------------------------------------------- utilities
__global__ void zero_cnt_kernel(int* cnt) {
    if (threadIdx.x < EM) cnt[threadIdx.x] = 0;
}

// fp32 [R][C] -> bf16 [C][R]  (LDS-tiled transpose+convert)
__global__ __launch_bounds__(256)
void transpose_cvt_kernel(const float* __restrict__ src, __bf16* __restrict__ dst,
                          int R, int C) {
    __shared__ float tile[32][33];
    int c0 = blockIdx.x * 32, r0 = blockIdx.y * 32;
    int tx = threadIdx.x, ty = threadIdx.y;           // block (32,8)
    #pragma unroll
    for (int i = ty; i < 32; i += 8)
        tile[i][tx] = src[(size_t)(r0 + i) * C + c0 + tx];
    __syncthreads();
    #pragma unroll
    for (int i = ty; i < 32; i += 8)
        dst[(size_t)(c0 + i) * R + r0 + tx] = to_bf16(tile[tx][i]);
}

// ------------------------------------------------------------------- router
// One wave32 per token: coalesced x reads, shuffle reduction, top-2 softmax,
// append (token*2+k) slot to the owning expert's list.
__global__ __launch_bounds__(256)
void router_kernel(const float* __restrict__ x, const float* __restrict__ Wr,
                   const float* __restrict__ br, float* __restrict__ wk,
                   int* __restrict__ lst, int* __restrict__ cnt) {
    int wave = threadIdx.x >> 5, lane = threadIdx.x & 31;
    int t = blockIdx.x * 8 + wave;
    const float* xr = x + (size_t)t * DM;
    float acc[EM];
    #pragma unroll
    for (int e = 0; e < EM; ++e) acc[e] = 0.f;
    for (int d = lane; d < DM; d += 32) {
        float xv = xr[d];
        const float* wrow = Wr + (size_t)d * EM;
        #pragma unroll
        for (int e = 0; e < EM; ++e) acc[e] += xv * wrow[e];
    }
    #pragma unroll
    for (int off = 16; off >= 1; off >>= 1)
        #pragma unroll
        for (int e = 0; e < EM; ++e) acc[e] += __shfl_down(acc[e], off, 32);

    if (lane == 0) {
        float lg[EM];
        #pragma unroll
        for (int e = 0; e < EM; ++e) lg[e] = acc[e] + br[e];
        int e0 = 0; float v0 = lg[0];
        #pragma unroll
        for (int e = 1; e < EM; ++e) if (lg[e] > v0) { v0 = lg[e]; e0 = e; }
        int e1 = -1; float v1 = -3.4e38f;
        #pragma unroll
        for (int e = 0; e < EM; ++e) if (e != e0 && lg[e] > v1) { v1 = lg[e]; e1 = e; }
        float p1 = __expf(v1 - v0);
        float inv = __builtin_amdgcn_rcpf(1.f + p1);
        wk[2 * t + 0] = inv;          // softmax over (v0, v1)
        wk[2 * t + 1] = p1 * inv;
        int p0 = atomicAdd(&cnt[e0], 1);
        lst[e0 * CAP + p0] = 2 * t + 0;
        int pp = atomicAdd(&cnt[e1], 1);
        lst[e1 * CAP + pp] = 2 * t + 1;
    }
}

// -------------------------------------------------- pass A: gate/up + SwiGLU
// grid.x = expert*(CAP/MT) + token-tile, grid.y = H/NT. h_buf[slot][HM] bf16.
__global__ __launch_bounds__(256)
void gateup_kernel(const float* __restrict__ x,
                   const __bf16* __restrict__ WgT, const __bf16* __restrict__ WuT,
                   const float* __restrict__ bg, const float* __restrict__ bu,
                   const int* __restrict__ lst, const int* __restrict__ cnt,
                   __bf16* __restrict__ h_buf) {
    int e = blockIdx.x >> MSH;
    int m_base = (blockIdx.x & ((CAP / MT) - 1)) * MT;
    int count = cnt[e];
    if (m_base >= count) return;
    int n0blk = blockIdx.y * NT;

    __shared__ __attribute__((aligned(16))) __bf16 xs[MT][LDA];
    __shared__ int slot_s[MT];
    int tid = threadIdx.x;
    if (tid < MT) {
        int pos = m_base + tid;
        slot_s[tid] = (pos < count) ? lst[e * CAP + pos] : -1;
    }
    __syncthreads();

    int wave = tid >> 5, lane = tid & 31;
    int n_wave = n0blk + wave * 16;
    const __bf16* Bg = WgT + (size_t)e * HM * DM;   // [H][D] (transposed)
    const __bf16* Bu = WuT + (size_t)e * HM * DM;
    int bn   = n_wave + (lane & 15);
    int koff = (lane >> 4) * 16;                    // 0 or 16
    int arow = lane & 15;

    v8f accg[8] = {}; v8f accu[8] = {};

    int srow = tid >> 1;            // 128 rows, 2 threads per row
    int scol = (tid & 1) * 16;      // 16 floats each

    for (int k0 = 0; k0 < DM; k0 += KT) {
        __syncthreads();
        {   // stage gathered x rows (fp32 -> bf16) into LDS
            int slot  = slot_s[srow];
            int token = (slot >= 0) ? (slot >> 1) : 0;
            const float* xr = x + (size_t)token * DM + k0 + scol;
            float4 f[4];
            f[0] = *(const float4*)xr;       f[1] = *(const float4*)(xr + 4);
            f[2] = *(const float4*)(xr + 8); f[3] = *(const float4*)(xr + 12);
            __bf16* dp = &xs[srow][scol];
            #pragma unroll
            for (int j = 0; j < 4; ++j) {
                dp[4 * j + 0] = to_bf16(f[j].x); dp[4 * j + 1] = to_bf16(f[j].y);
                dp[4 * j + 2] = to_bf16(f[j].z); dp[4 * j + 3] = to_bf16(f[j].w);
            }
        }
        __syncthreads();

        FragAB fg, fu;
        const __bf16* bgp = Bg + (size_t)bn * DM + k0 + koff;
        const __bf16* bup = Bu + (size_t)bn * DM + k0 + koff;
        __builtin_prefetch(bgp + KT, 0, 3);          // global_prefetch_b8 (near)
        __builtin_prefetch(bup + KT, 0, 3);
        fg.q[0] = *(const uint4*)bgp; fg.q[1] = *(const uint4*)(bgp + 8);
        fu.q[0] = *(const uint4*)bup; fu.q[1] = *(const uint4*)(bup + 8);

        #pragma unroll
        for (int mi = 0; mi < 8; ++mi) {
            FragAB a;
            const __bf16* ap = &xs[mi * 16 + arow][koff];
            a.q[0] = *(const uint4*)ap; a.q[1] = *(const uint4*)(ap + 8);
            accg[mi] = wmma_bf16(a.v, fg.v, accg[mi]);
            accu[mi] = wmma_bf16(a.v, fu.v, accu[mi]);
        }
    }

    // epilogue: bias + SwiGLU (v_rcp_f32 sigmoid), store bf16 h rows
    int col = n_wave + (lane & 15);
    float bgv = bg[e * HM + col], buv = bu[e * HM + col];
    int rowoff = (lane < 16) ? 0 : 8;
    #pragma unroll
    for (int mi = 0; mi < 8; ++mi) {
        #pragma unroll
        for (int r = 0; r < 8; ++r) {
            int m = mi * 16 + r + rowoff;
            int pos = m_base + m;
            if (pos < count) {
                float g = accg[mi][r] + bgv;
                float u = accu[mi][r] + buv;
                float s = __builtin_amdgcn_rcpf(1.f + __expf(-g));
                float h = u * g * s;                  // u * g * sigmoid(g)
                int slot = slot_s[m];
                h_buf[(size_t)slot * HM + col] = to_bf16(h);
            }
        }
    }
}

// ------------------------------------------------ pass B: down-proj + scale
// grid.x = expert*(CAP/MT) + token-tile, grid.y = D/NT. y_slot[slot][DM] f32.
__global__ __launch_bounds__(256)
void down_kernel(const __bf16* __restrict__ h_buf, const __bf16* __restrict__ WdT,
                 const float* __restrict__ bd, const float* __restrict__ wk,
                 const int* __restrict__ lst, const int* __restrict__ cnt,
                 float* __restrict__ y_slot) {
    int e = blockIdx.x >> MSH;
    int m_base = (blockIdx.x & ((CAP / MT) - 1)) * MT;
    int count = cnt[e];
    if (m_base >= count) return;
    int n0blk = blockIdx.y * NT;

    __shared__ __attribute__((aligned(16))) __bf16 hs[MT][LDA];
    __shared__ int slot_s[MT];
    int tid = threadIdx.x;
    if (tid < MT) {
        int pos = m_base + tid;
        slot_s[tid] = (pos < count) ? lst[e * CAP + pos] : -1;
    }
    __syncthreads();

    int wave = tid >> 5, lane = tid & 31;
    int n_wave = n0blk + wave * 16;
    const __bf16* Bd = WdT + (size_t)e * DM * HM;    // [D][H] (transposed)
    int bn   = n_wave + (lane & 15);
    int koff = (lane >> 4) * 16;
    int arow = lane & 15;

    v8f acc[8] = {};
    int srow = tid >> 1;
    int scol = (tid & 1) * 16;

    for (int k0 = 0; k0 < HM; k0 += KT) {
        __syncthreads();
        {   // stage gathered h rows (already bf16): 32B per thread
            int slot = slot_s[srow];
            int hsrc = (slot >= 0) ? slot : 0;
            const __bf16* gp = h_buf + (size_t)hsrc * HM + k0 + scol;
            __bf16* lp = &hs[srow][scol];
#ifdef USE_ASYNC_LDS
            // async global->LDS DMA (ASYNCcnt path), 16B x2 per thread
            __builtin_amdgcn_load_to_lds(
                (void*)gp,
                (__attribute__((address_space(3))) void*)(unsigned)(uintptr_t)lp,
                16, 0, 0);
            __builtin_amdgcn_load_to_lds(
                (void*)(gp + 8),
                (__attribute__((address_space(3))) void*)(unsigned)(uintptr_t)(lp + 8),
                16, 0, 0);
#else
            *(uint4*)lp       = *(const uint4*)gp;
            *(uint4*)(lp + 8) = *(const uint4*)(gp + 8);
#endif
        }
#ifdef USE_ASYNC_LDS
        __builtin_amdgcn_s_wait_asynccnt(0);
#endif
        __syncthreads();

        FragAB fb;
        const __bf16* bdp = Bd + (size_t)bn * HM + k0 + koff;
        __builtin_prefetch(bdp + KT, 0, 3);
        fb.q[0] = *(const uint4*)bdp; fb.q[1] = *(const uint4*)(bdp + 8);

        #pragma unroll
        for (int mi = 0; mi < 8; ++mi) {
            FragAB a;
            const __bf16* ap = &hs[mi * 16 + arow][koff];
            a.q[0] = *(const uint4*)ap; a.q[1] = *(const uint4*)(ap + 8);
            acc[mi] = wmma_bf16(a.v, fb.v, acc[mi]);
        }
    }

    int col = n_wave + (lane & 15);
    float bdv = bd[e * DM + col];
    int rowoff = (lane < 16) ? 0 : 8;
    #pragma unroll
    for (int mi = 0; mi < 8; ++mi) {
        #pragma unroll
        for (int r = 0; r < 8; ++r) {
            int m = mi * 16 + r + rowoff;
            int pos = m_base + m;
            if (pos < count) {
                int slot = slot_s[m];
                float w = wk[slot];
                y_slot[(size_t)slot * DM + col] = w * (acc[mi][r] + bdv);
            }
        }
    }
}

// ----------------------------------------------- pass C: combine two slots
__global__ __launch_bounds__(256)
void combine_kernel(const float* __restrict__ y_slot, float* __restrict__ out) {
    int i4 = blockIdx.x * blockDim.x + threadIdx.x;     // float4 index
    int token = i4 / (DM / 4);
    int c     = (i4 % (DM / 4)) * 4;
    const float4 a = *(const float4*)(y_slot + (size_t)(2 * token + 0) * DM + c);
    const float4 b = *(const float4*)(y_slot + (size_t)(2 * token + 1) * DM + c);
    float4 o; o.x = a.x + b.x; o.y = a.y + b.y; o.z = a.z + b.z; o.w = a.w + b.w;
    *(float4*)(out + (size_t)token * DM + c) = o;
}

// ---------------------------------------------------------------------------
extern "C" void kernel_launch(void* const* d_in, const int* in_sizes, int n_in,
                              void* d_out, int out_size, void* d_ws, size_t ws_size,
                              hipStream_t stream) {
    const float* x  = (const float*)d_in[0];
    const float* Wr = (const float*)d_in[1];
    const float* br = (const float*)d_in[2];
    const float* Wg = (const float*)d_in[3];
    const float* bg = (const float*)d_in[4];
    const float* Wu = (const float*)d_in[5];
    const float* bu = (const float*)d_in[6];
    const float* Wd = (const float*)d_in[7];
    const float* bd = (const float*)d_in[8];
    float* out = (float*)d_out;

    // workspace layout
    char* ws = (char*)d_ws;
    size_t off = 0;
    __bf16* WgT   = (__bf16*)(ws + off); off += (size_t)EM * HM * DM * 2;   // 67MB
    __bf16* WuT   = (__bf16*)(ws + off); off += (size_t)EM * HM * DM * 2;   // 67MB
    __bf16* WdT   = (__bf16*)(ws + off); off += (size_t)EM * DM * HM * 2;   // 67MB
    __bf16* h_buf = (__bf16*)(ws + off); off += (size_t)2 * TKN * HM * 2;   // 134MB
    float*  y_slot= (float*) (ws + off); off += (size_t)2 * TKN * DM * 4;   // 67MB
    float*  wk    = (float*) (ws + off); off += (size_t)2 * TKN * 4;
    int*    lst   = (int*)   (ws + off); off += (size_t)EM * CAP * 4;
    int*    cnt   = (int*)   (ws + off); off += 256;

    zero_cnt_kernel<<<1, 32, 0, stream>>>(cnt);
    router_kernel<<<TKN / 8, 256, 0, stream>>>(x, Wr, br, wk, lst, cnt);

    // weight convert+transpose: Wg/Wu [D][H] -> [H][D], Wd [H][D] -> [D][H]
    for (int e = 0; e < EM; ++e) {
        transpose_cvt_kernel<<<dim3(HM / 32, DM / 32), dim3(32, 8), 0, stream>>>(
            Wg + (size_t)e * DM * HM, WgT + (size_t)e * HM * DM, DM, HM);
        transpose_cvt_kernel<<<dim3(HM / 32, DM / 32), dim3(32, 8), 0, stream>>>(
            Wu + (size_t)e * DM * HM, WuT + (size_t)e * HM * DM, DM, HM);
        transpose_cvt_kernel<<<dim3(DM / 32, HM / 32), dim3(32, 8), 0, stream>>>(
            Wd + (size_t)e * HM * DM, WdT + (size_t)e * DM * HM, HM, DM);
    }

    gateup_kernel<<<dim3(EM * (CAP / MT), HM / NT), 256, 0, stream>>>(
        x, WgT, WuT, bg, bu, lst, cnt, h_buf);
    down_kernel<<<dim3(EM * (CAP / MT), DM / NT), 256, 0, stream>>>(
        h_buf, WdT, bd, wk, lst, cnt, y_slot);
    combine_kernel<<<(TKN * DM / 4) / 256, 256, 0, stream>>>(y_slot, out);
}
